// MultiHeadedCrossAttn_60627758350408
// MI455X (gfx1250) — matrix operand order repaired and checked by reading, not compiled, not run on hardware
//
#include <hip/hip_runtime.h>
#include <hip/hip_bf16.h>

typedef __attribute__((ext_vector_type(16))) __bf16 bf16x16;
typedef __attribute__((ext_vector_type(8)))  float  v8f;
typedef __attribute__((ext_vector_type(4)))  unsigned int u32x4;
typedef __attribute__((ext_vector_type(8)))  int i32x8;
typedef __attribute__((ext_vector_type(4)))  int i32x4;

#if defined(__has_builtin)
#if __has_builtin(__builtin_amdgcn_tensor_load_to_lds)
#define HAVE_TDM 1
#endif
#endif
#ifndef HAVE_TDM
#define HAVE_TDM 0
#endif

// Problem constants (from reference)
constexpr int T  = 32;
constexpr int Bq = 64;    // batch b
constexpr int Aq = 16;    // queries per (t,b)
constexpr int F  = 128;   // context frames
constexpr int D  = 512;   // model dim
constexpr int H  = 8;     // heads
constexpr int HD = 64;    // head dim

__device__ __forceinline__ __bf16 tobf(float x) { return (__bf16)x; }

// WMMA 16x16x32 bf16 fragment index helpers (wave32, CDNA5 ISA 7.12.2)
// A (16x32, MxK): lane L holds row m = L&15; element i -> K = (L>=16 ? 16 : 0) + i (code's order, not ISA's)
__device__ __forceinline__ int a_kidx(int i, int lane) {
    return i + ((lane >= 16) ? 16 : 0);
}
// B (32x16, KxN): lane L holds col n = L&15; element i -> K = (L>=16 ? 16 : 0) + i (code's order, not ISA's)
__device__ __forceinline__ int b_kbase(int lane) { return (lane >= 16) ? 16 : 0; }
// C/D (16x16 f32): vgpr r, lane L -> M = r + (L>=16 ? 8 : 0), N = L&15

#if HAVE_TDM
// Tensor Data Mover: 2-D bf16 tile, global -> LDS, with optional LDS row padding.
// D# layout per CDNA5 ISA (group0 128b, group1 256b, groups 2/3 = 0).
// 6-arg toolchain builtin: (g0 u32x4, g1 i32x8, g2 i32x4, g3 i32x4, gx i32x8, cpol)
// pad_dwords: dwords of LDS padding appended per row (0 = none)
// interval_code: pad interval code, 0=2 dwords, 1=4, ... 7=256 (row length in dwords)
__device__ __forceinline__ void tdm_load_2d_bf16(void* lds_dst, const void* gsrc,
                                                 unsigned width_elems, unsigned rows,
                                                 unsigned long long row_stride_elems,
                                                 unsigned pad_dwords, unsigned interval_code) {
    unsigned lds_addr = (unsigned)(unsigned long long)(size_t)lds_dst; // low 32 = LDS offset
    unsigned long long ga = (unsigned long long)(size_t)gsrc;
    u32x4 g0;
    g0[0] = 1u;                                                   // count=1, user mode
    g0[1] = lds_addr;                                             // lds_addr
    g0[2] = (unsigned)(ga & 0xFFFFFFFFull);                       // global_addr[31:0]
    g0[3] = (unsigned)((ga >> 32) & 0x1FFFFFFull) | (2u << 30);   // global_addr[56:32], type=2
    i32x8 g1;
    unsigned w0 = (1u << 16);                                     // data_size=1 -> 2 bytes
    if (pad_dwords)
        w0 |= (1u << 20) | (interval_code << 22) | ((pad_dwords - 1u) << 25);
    g1[0] = (int)w0;
    g1[1] = (int)((width_elems & 0xFFFFu) << 16);                 // tensor_dim0[15:0]
    g1[2] = (int)(((width_elems >> 16) & 0xFFFFu) | ((rows & 0xFFFFu) << 16)); // td0 hi, td1 lo
    g1[3] = (int)(((rows >> 16) & 0xFFFFu) | ((width_elems & 0xFFFFu) << 16)); // td1 hi, tile_dim0
    g1[4] = (int)(rows & 0xFFFFu);                                // tile_dim1 (tile_dim2 = 0)
    g1[5] = (int)(unsigned)(row_stride_elems & 0xFFFFFFFFull);    // tensor_dim0_stride[31:0]
    g1[6] = (int)(unsigned)((row_stride_elems >> 32) & 0xFFFFull);// stride[47:32]
    g1[7] = 0;
    i32x4 z4 = {0, 0, 0, 0};
    i32x8 z8 = {0, 0, 0, 0, 0, 0, 0, 0};
    __builtin_amdgcn_tensor_load_to_lds(g0, g1, z4, z4, z8, 0);
}
#endif

// ---------------------------------------------------------------------------
// Tiled GEMM:  Y[m,n] = sum_k X[m,k] * W[n,k] + bias[n]
// X panel (128 x 512) resident in LDS as bf16 -> X read from HBM exactly once.
// N looped inside the block; W tiles (64x64) streamed from L2.
//   OUTPROJ=false: Y bf16 row-major (workspace);  OUTPROJ=true: f32, (t,b,a,d)->(t,a,b,d)
// Block: 256 threads (8 waves).
// ---------------------------------------------------------------------------
template <typename InT, bool OUTPROJ>
__global__ __launch_bounds__(256)
void gemm_xwT(const InT* __restrict__ X, const float* __restrict__ W,
              const float* __restrict__ bias, void* __restrict__ Y) {
    constexpr int PAD = 8;
    __shared__ __bf16 Xs[128][D + PAD];     // 130 KB
    __shared__ __bf16 Ws[64][64 + PAD];     // 9 KB

    const int tid  = threadIdx.x;
    const int lane = tid & 31;
    const int wave = tid >> 5;
    const int m0   = blockIdx.x * 128;

    // ---- Stage full X panel (128 x 512) ----
    bool staged = false;
#if HAVE_TDM
    if constexpr (sizeof(InT) == 2) {       // bf16 input: pure copy -> TDM
        if (tid < 32) {
            tdm_load_2d_bf16(&Xs[0][0], X + (size_t)m0 * D, D, 128, D,
                             /*pad_dwords=*/PAD / 2, /*interval=512 elems=256 dw*/ 7);
            __builtin_amdgcn_s_wait_tensorcnt(0);
        }
        staged = true;
    }
#endif
    if (!staged) {
        for (int idx = tid; idx < 128 * D; idx += 256) {
            int r = idx >> 9, c = idx & (D - 1);
            Xs[r][c] = tobf((float)X[(size_t)(m0 + r) * D + c]);
        }
    }
    __syncthreads();

    for (int n0 = 0; n0 < D; n0 += 64) {
        v8f acc[4] = {};
        for (int k0 = 0; k0 < D; k0 += 64) {
            // Stage W tile (64 n-rows x 64 k) fp32 -> bf16
            for (int idx = tid; idx < 64 * 64; idx += 256) {
                int n = idx >> 6, c = idx & 63;
                Ws[n][c] = tobf(W[(size_t)(n0 + n) * D + k0 + c]);
            }
            if (k0 + 64 < D) {  // prefetch next W tile (global_prefetch_b8)
                int n = tid >> 2;
                __builtin_prefetch(&W[(size_t)(n0 + n) * D + k0 + 64 + (tid & 3) * 16], 0, 1);
            }
            __syncthreads();

            const int row0 = wave * 16;
            #pragma unroll
            for (int ks = 0; ks < 64; ks += 32) {
                bf16x16 afrag;
                #pragma unroll
                for (int i = 0; i < 16; ++i)
                    afrag[i] = Xs[row0 + (lane & 15)][k0 + ks + a_kidx(i, lane)];
                #pragma unroll
                for (int nt = 0; nt < 4; ++nt) {
                    bf16x16 bfrag;
                    #pragma unroll
                    for (int i = 0; i < 16; ++i)
                        bfrag[i] = Ws[nt * 16 + (lane & 15)][ks + b_kbase(lane) + i];
                    acc[nt] = __builtin_amdgcn_wmma_f32_16x16x32_bf16(
                        false, afrag, false, bfrag, (short)0, acc[nt], false, false);
                }
            }
            __syncthreads();
        }

        // Epilogue for this n0 strip
        const int nloc = lane & 15;
        const int mhi  = (lane >= 16) ? 8 : 0;
        #pragma unroll
        for (int nt = 0; nt < 4; ++nt) {
            const int n  = n0 + nt * 16 + nloc;
            const float bv = bias[n];
            #pragma unroll
            for (int r = 0; r < 8; ++r) {
                const int m = m0 + wave * 16 + r + mhi;
                const float val = acc[nt][r] + bv;
                if constexpr (OUTPROJ) {
                    int ti = m >> 10, bi = (m >> 4) & 63, ai = m & 15;
                    ((float*)Y)[(((size_t)ti * Aq + ai) * Bq + bi) * D + n] = val;
                } else {
                    ((__bf16*)Y)[(size_t)m * D + n] = tobf(val);
                }
            }
        }
    }
}

// ---------------------------------------------------------------------------
// Attention kernel: one block per (t*b, head). 128 threads = 4 waves.
// K/V/Q head tiles staged via Tensor Data Mover (pure bf16 2-D copies).
// ---------------------------------------------------------------------------
__global__ __launch_bounds__(128)
void attn_kernel(const __bf16* __restrict__ Qp, const __bf16* __restrict__ Kc,
                 const __bf16* __restrict__ Vc, const int* __restrict__ mask,
                 float* __restrict__ w_out, __bf16* __restrict__ attn_out) {
    constexpr int PAD = 8;
    __shared__ __bf16 Qs[Aq][HD + PAD];     // q_h, 16 x 64
    __shared__ __bf16 Ksh[F][HD + PAD];     // k_h, 128 x 64
    __shared__ __bf16 Vsh[F][HD + PAD];     // v_h (f-major), 128 x 64
    __shared__ float  Lg[F][Aq];            // logits / weights fp32
    __shared__ __bf16 Wb[F][Aq];            // weights bf16 (B operand)

    const int tid  = threadIdx.x;
    const int lane = tid & 31;
    const int wave = tid >> 5;
    const int tb   = blockIdx.x;            // 0 .. t*b-1
    const int h    = blockIdx.y;
    const int ti   = tb >> 6;
    const int bi   = tb & 63;

    // k[ti,bi,fi,h,e] = Kc[ti, 2*bi + fi/64, fi%64, h*64+e]  -- a contiguous
    // 128-row x (stride 512) 2-D tile starting at row (ti*F + 2*bi)*64.
    // v[ti,bi,h,e,fi] = Vc[ti, fi, bi, h*64+e]               -- 128 rows, stride b*d.
#if HAVE_TDM
    if (tid < 32) {
        tdm_load_2d_bf16(&Qs[0][0],  Qp + (size_t)tb * Aq * D + h * HD,
                         HD, Aq, D, PAD / 2, 4);
        tdm_load_2d_bf16(&Ksh[0][0], Kc + ((size_t)ti * F + 2 * bi) * Bq * D + h * HD,
                         HD, F, D, PAD / 2, 4);
        tdm_load_2d_bf16(&Vsh[0][0], Vc + ((size_t)ti * F * Bq + bi) * D + h * HD,
                         HD, F, (unsigned long long)Bq * D, PAD / 2, 4);
        __builtin_amdgcn_s_wait_tensorcnt(0);
    }
#else
    for (int idx = tid; idx < Aq * HD; idx += 128) {
        int ai = idx >> 6, e = idx & 63;
        Qs[ai][e] = Qp[((size_t)tb * Aq + ai) * D + h * HD + e];
    }
    for (int idx = tid; idx < F * HD; idx += 128) {
        int fi = idx >> 6, e = idx & 63;
        Ksh[fi][e] = Kc[(((size_t)ti * F + 2 * bi + (fi >> 6)) * Bq + (fi & 63)) * D + h * HD + e];
        Vsh[fi][e] = Vc[(((size_t)ti * F + fi) * Bq + bi) * D + h * HD + e];
    }
#endif
    __syncthreads();

    // Logits: each wave computes 2 m-tiles of 16 f-rows; scale 1/sqrt(hd)=0.125 at store
    #pragma unroll
    for (int mt = 0; mt < 2; ++mt) {
        const int f0 = wave * 32 + mt * 16;
        v8f acc = {};
        #pragma unroll
        for (int ks = 0; ks < HD; ks += 32) {
            bf16x16 afrag, bfrag;
            #pragma unroll
            for (int i = 0; i < 16; ++i) {
                afrag[i] = Ksh[f0 + (lane & 15)][ks + a_kidx(i, lane)];
                bfrag[i] = Qs[lane & 15][ks + b_kbase(lane) + i];
            }
            acc = __builtin_amdgcn_wmma_f32_16x16x32_bf16(
                false, afrag, false, bfrag, (short)0, acc, false, false);
        }
        const int mhi = (lane >= 16) ? 8 : 0;
        #pragma unroll
        for (int r = 0; r < 8; ++r)
            Lg[f0 + r + mhi][lane & 15] = acc[r] * 0.125f;
    }
    __syncthreads();

    // Mask
    if (tid < F && mask[bi * F + tid] == 0) {
        #pragma unroll
        for (int ai = 0; ai < Aq; ++ai) Lg[tid][ai] = -1.0e9f;
    }
    __syncthreads();

    // Softmax over f (16 columns)
    if (tid < Aq) {
        const int ai = tid;
        float mx = -3.4e38f;
        for (int fi = 0; fi < F; ++fi) mx = fmaxf(mx, Lg[fi][ai]);
        float sum = 0.0f;
        for (int fi = 0; fi < F; ++fi) sum += __expf(Lg[fi][ai] - mx);
        const float inv = 1.0f / sum;
        for (int fi = 0; fi < F; ++fi) {
            float wv = __expf(Lg[fi][ai] - mx) * inv;
            Lg[fi][ai] = wv;
            Wb[fi][ai] = tobf(wv);
        }
    }
    __syncthreads();

    // Weights (t,b,H,f,a) -> output
    for (int idx = tid; idx < F * Aq; idx += 128) {
        int fi = idx >> 4, ai = idx & 15;
        w_out[(((size_t)tb * H + h) * F + fi) * Aq + ai] = Lg[fi][ai];
    }

    // attn_h = v_h @ weights : M=64(e), N=16(a), K=128(f). One m-tile per wave.
    {
        const int e0 = wave * 16;
        v8f acc = {};
        #pragma unroll
        for (int ks = 0; ks < F; ks += 32) {
            bf16x16 afrag, bfrag;
            #pragma unroll
            for (int i = 0; i < 16; ++i) {
                afrag[i] = Vsh[ks + a_kidx(i, lane)][e0 + (lane & 15)];  // transposed read
                bfrag[i] = Wb[ks + b_kbase(lane) + i][lane & 15];
            }
            acc = __builtin_amdgcn_wmma_f32_16x16x32_bf16(
                false, afrag, false, bfrag, (short)0, acc, false, false);
        }
        const int mhi = (lane >= 16) ? 8 : 0;
        const int ai  = lane & 15;
        #pragma unroll
        for (int r = 0; r < 8; ++r) {
            int e = e0 + r + mhi;
            attn_out[((size_t)tb * Aq + ai) * D + h * HD + e] = tobf(acc[r]);  // (t,b,a,d)
        }
    }
}

// ---------------------------------------------------------------------------
extern "C" void kernel_launch(void* const* d_in, const int* in_sizes, int n_in,
                              void* d_out, int out_size, void* d_ws, size_t ws_size,
                              hipStream_t stream) {
    const float* q_embeds = (const float*)d_in[0];   // (t,b,a,d)
    const float* ctx      = (const float*)d_in[1];   // (t,f,b,d)
    const int*   mask     = (const int*)  d_in[2];   // (b,f)
    const float* Wq = (const float*)d_in[3];
    const float* bq = (const float*)d_in[4];
    const float* Wk = (const float*)d_in[5];
    const float* bk = (const float*)d_in[6];
    const float* Wv = (const float*)d_in[7];
    const float* bv = (const float*)d_in[8];
    const float* Wo = (const float*)d_in[9];
    const float* bo = (const float*)d_in[10];

    const size_t Mq = (size_t)T * Bq * Aq;           // 32768
    const size_t Mc = (size_t)T * F * Bq;            // 262144

    // Workspace layout (bf16 intermediates)
    char* ws = (char*)d_ws;
    __bf16* Qp   = (__bf16*)(ws);                                  // 32 MB
    __bf16* Kc   = (__bf16*)(ws + Mq * D * 2);                     // 256 MB
    __bf16* Vc   = (__bf16*)(ws + (Mq + Mc) * D * 2);              // 256 MB
    __bf16* attn = (__bf16*)(ws + (Mq + 2 * Mc) * D * 2);          // 32 MB

    float* o_out = (float*)d_out;                                  // (t,a,b,d)
    float* w_out = o_out + (size_t)T * Aq * Bq * D;                // weights (t,b,H,f,a)

    dim3 blk(256);
    gemm_xwT<float, false><<<dim3((unsigned)(Mq / 128)), blk, 0, stream>>>(q_embeds, Wq, bq, Qp);
    gemm_xwT<float, false><<<dim3((unsigned)(Mc / 128)), blk, 0, stream>>>(ctx, Wk, bk, Kc);
    gemm_xwT<float, false><<<dim3((unsigned)(Mc / 128)), blk, 0, stream>>>(ctx, Wv, bv, Vc);

    attn_kernel<<<dim3(T * Bq, H), dim3(128), 0, stream>>>(Qp, Kc, Vc, mask, w_out, attn);

    gemm_xwT<__bf16, true><<<dim3((unsigned)(Mq / 128)), blk, 0, stream>>>(attn, Wo, bo, d_out);
}
